// EfficientViMBlock_71992241816069
// MI455X (gfx1250) — compile-verified, matrix operand order checked by reading, not run on hardware
//
#include <hip/hip_runtime.h>
#include <hip/hip_bf16.h>
#include <math.h>

typedef __bf16 bf16_t;
typedef __attribute__((ext_vector_type(16))) __bf16 v16bf;
typedef __attribute__((ext_vector_type(8)))  __bf16 v8bf;
typedef __attribute__((ext_vector_type(8)))  float  v8f;

#define DEVI __device__ __forceinline__

static constexpr int   Bn  = 16;
static constexpr int   Cd  = 256;    // DIM
static constexpr int   Ll  = 4096;   // 64*64
static constexpr int   St  = 64;     // STATE
static constexpr int   O3  = 192;    // 3*STATE
static constexpr int   HID = 1024;
static constexpr float EPS = 1e-5f;

DEVI float sigm_(float x) { return 1.f / (1.f + __expf(-x)); }

// ---------------------------------------------------------------------------
// fp32 -> bf16 flat convert (weights, ho)
// ---------------------------------------------------------------------------
__global__ void k_cvt_bf16(const float* __restrict__ s, bf16_t* __restrict__ d, int n) {
  int i = blockIdx.x * 256 + threadIdx.x;
  if (i < n) d[i] = (bf16_t)s[i];
}

// ---------------------------------------------------------------------------
// depthwise 3x3, pad 1, NCHW fp32. grid.x = Cc*Ll/256, grid.z = batch
// ---------------------------------------------------------------------------
__global__ void k_dwconv3x3(const float* __restrict__ x, const float* __restrict__ w,
                            const float* __restrict__ bias, float* __restrict__ out, int Cc) {
  int idx = blockIdx.x * 256 + threadIdx.x;      // c*4096 + h*64 + w
  int b = blockIdx.z;
  int c = idx >> 12;
  int hw = idx & 4095;
  int h = hw >> 6, wx = hw & 63;
  const float* xp = x + ((size_t)b * Cc + c) * (size_t)Ll;
  const float* wp = w + c * 9;
  float acc = bias ? bias[c] : 0.f;
  #pragma unroll
  for (int dy = -1; dy <= 1; ++dy) {
    int hh = h + dy;
    if (hh < 0 || hh > 63) continue;
    #pragma unroll
    for (int dx = -1; dx <= 1; ++dx) {
      int ww = wx + dx;
      if (ww < 0 || ww > 63) continue;
      acc += wp[(dy + 1) * 3 + (dx + 1)] * xp[hh * 64 + ww];
    }
  }
  out[((size_t)b * Cc + c) * (size_t)Ll + hw] = acc;
}

// ---------------------------------------------------------------------------
// per-pixel channel mean/rstd over NCHW fp32. grid.x = Bn*Ll/256
// ---------------------------------------------------------------------------
__global__ void k_chan_stats(const float* __restrict__ src, float* __restrict__ mean,
                             float* __restrict__ rstd, int Cc) {
  int g = blockIdx.x * 256 + threadIdx.x;  // b*4096 + l
  int b = g >> 12, l = g & 4095;
  const float* p = src + (size_t)b * Cc * (size_t)Ll + l;
  float s1 = 0.f, s2 = 0.f;
  for (int c = 0; c < Cc; ++c) {
    float v = p[(size_t)c * Ll];
    s1 += v; s2 += v * v;
  }
  float m = s1 / (float)Cc;
  float var = s2 / (float)Cc - m * m;
  mean[g] = m;
  rstd[g] = rsqrtf(var + EPS);
}

// ---------------------------------------------------------------------------
// out = (1-sig(al[c]))*xin + sig(al[c])*(lw[c]*(conv-mean)*rstd + lb[c])
// grid.x = Cd*Ll/256, grid.z = batch
// ---------------------------------------------------------------------------
__global__ void k_ln_gate(const float* __restrict__ xin, const float* __restrict__ conv,
                          const float* __restrict__ mean, const float* __restrict__ rstd,
                          const float* __restrict__ lw, const float* __restrict__ lb,
                          const float* __restrict__ al, float* __restrict__ out) {
  int idx = blockIdx.x * 256 + threadIdx.x;
  int b = blockIdx.z;
  int c = idx >> 12, l = idx & 4095;
  size_t off = ((size_t)b * Cd + c) * (size_t)Ll + l;
  int pg = b * Ll + l;
  float a = sigm_(al[c]);
  float n = lw[c] * (conv[off] - mean[pg]) * rstd[pg] + lb[c];
  out[off] = (1.f - a) * xin[off] + a * n;
}

// ---------------------------------------------------------------------------
// xn = lw*(x1-mean)*rstd+lb  -> fp32 + bf16 (same [c][l] layout)
// ---------------------------------------------------------------------------
__global__ void k_ln_pack(const float* __restrict__ x1, const float* __restrict__ mean,
                          const float* __restrict__ rstd, const float* __restrict__ lw,
                          const float* __restrict__ lb, float* __restrict__ of,
                          bf16_t* __restrict__ ob) {
  int idx = blockIdx.x * 256 + threadIdx.x;
  int b = blockIdx.z;
  int c = idx >> 12, l = idx & 4095;
  size_t off = ((size_t)b * Cd + c) * (size_t)Ll + l;
  int pg = b * Ll + l;
  float n = lw[c] * (x1[off] - mean[pg]) * rstd[pg] + lb[c];
  of[off] = n;
  ob[off] = (bf16_t)n;
}

// ---------------------------------------------------------------------------
// fp32 [R][Cc] -> bf16 [Cc][R], LDS 32x33 tiles. grid(Cc/32, R/32, Bn)
// ---------------------------------------------------------------------------
__global__ void k_transpack(const float* __restrict__ src, bf16_t* __restrict__ dst,
                            int R, int Cc, size_t sbs, size_t dbs) {
  __shared__ float tile[32][33];
  int b = blockIdx.z;
  int r0 = blockIdx.y * 32, c0 = blockIdx.x * 32;
  int tx = threadIdx.x & 31, ty = threadIdx.x >> 5;   // 256 threads: ty 0..7
  const float* s = src + (size_t)b * sbs;
  #pragma unroll
  for (int i = ty; i < 32; i += 8) tile[i][tx] = s[(size_t)(r0 + i) * Cc + c0 + tx];
  __syncthreads();
  bf16_t* d = dst + (size_t)b * dbs;
  #pragma unroll
  for (int i = ty; i < 32; i += 8) d[(size_t)(c0 + i) * R + r0 + tx] = (bf16_t)tile[tx][i];
}

// ---------------------------------------------------------------------------
// bf16 WMMA GEMM: C[M][N] = A[M][K] * Bt[N][K]^T (+ bias[m]).
// 128 threads = 4 waves, 2x2 wave grid, 32x32 per wave => 64x64 block tile.
// M,N % 64 == 0, K % 32 == 0. grid(N/64, M/64, batch)
// ---------------------------------------------------------------------------
DEVI v16bf ld_frag(const bf16_t* p, int o1, int o2) {
  v8bf x = *(const v8bf*)(p + o1);
  v8bf y = *(const v8bf*)(p + o2);
  return __builtin_shufflevector(x, y, 0, 1, 2, 3, 4, 5, 6, 7, 8, 9, 10, 11, 12, 13, 14, 15);
}

__global__ void __launch_bounds__(128)
k_gemm_bf16(const bf16_t* __restrict__ A, const bf16_t* __restrict__ Bt,
            float* __restrict__ C, const float* __restrict__ bias,
            int M, int N, int K, size_t asb, size_t bsb, size_t csb) {
  int b = blockIdx.z;
  const bf16_t* Ab = A + (size_t)b * asb;
  const bf16_t* Bb = Bt + (size_t)b * bsb;
  float* Cb = C + (size_t)b * csb;
  int wave = threadIdx.x >> 5;
  int lane = threadIdx.x & 31;
  int lo = lane & 15, hi = lane >> 4;
  int wm = wave >> 1, wn = wave & 1;
  int m0 = blockIdx.y * 64 + wm * 32;
  int n0 = blockIdx.x * 64 + wn * 32;
  v8f acc00 = {}, acc01 = {}, acc10 = {}, acc11 = {};
  const bf16_t* ar0 = Ab + (size_t)(m0 + lo) * K;
  const bf16_t* ar1 = Ab + (size_t)(m0 + 16 + lo) * K;
  const bf16_t* br0 = Bb + (size_t)(n0 + lo) * K;
  const bf16_t* br1 = Bb + (size_t)(n0 + 16 + lo) * K;
  const int k1 = hi * 8, k2 = 16 + hi * 8;     // ISA 16-bit A 16x32 per-lane K pattern
  for (int k0 = 0; k0 < K; k0 += 32) {
    v16bf a0 = ld_frag(ar0 + k0, k1, k2);
    v16bf a1 = ld_frag(ar1 + k0, k1, k2);
    v16bf b0 = ld_frag(br0 + k0, k1, k2);
    v16bf b1 = ld_frag(br1 + k0, k1, k2);
    __builtin_prefetch(ar0 + k0 + 64, 0, 1);
    __builtin_prefetch(br0 + k0 + 64, 0, 1);
    acc00 = __builtin_amdgcn_wmma_f32_16x16x32_bf16(false, a0, false, b0, (short)0, acc00, false, false);
    acc01 = __builtin_amdgcn_wmma_f32_16x16x32_bf16(false, a0, false, b1, (short)0, acc01, false, false);
    acc10 = __builtin_amdgcn_wmma_f32_16x16x32_bf16(false, a1, false, b0, (short)0, acc10, false, false);
    acc11 = __builtin_amdgcn_wmma_f32_16x16x32_bf16(false, a1, false, b1, (short)0, acc11, false, false);
  }
  // C/D layout: lanes 0-15: VGPR r -> (M=m0+r, N=lo); lanes 16-31: (M=m0+8+r, N=lo)
  #pragma unroll
  for (int r = 0; r < 8; ++r) {
    int row0 = m0 + hi * 8 + r;
    int row1 = m0 + 16 + hi * 8 + r;
    int col0 = n0 + lo;
    int col1 = n0 + 16 + lo;
    float bi0 = bias ? bias[row0] : 0.f;
    float bi1 = bias ? bias[row1] : 0.f;
    Cb[(size_t)row0 * N + col0] = acc00[r] + bi0;
    Cb[(size_t)row0 * N + col1] = acc01[r] + bi0;
    Cb[(size_t)row1 * N + col0] = acc10[r] + bi1;
    Cb[(size_t)row1 * N + col1] = acc11[r] + bi1;
  }
}

// ---------------------------------------------------------------------------
// softmax over L of (dt + A[s]); AB = softmax * Bm -> bf16 [b][s][l]
// block per (b,s); 256 threads x 16 elems in registers
// ---------------------------------------------------------------------------
__global__ void __launch_bounds__(256)
k_softmax_ab(const float* __restrict__ bcdt, const float* __restrict__ Av,
             bf16_t* __restrict__ ab) {
  __shared__ float red[8];
  __shared__ float bc;
  int b = blockIdx.z, s = blockIdx.x;
  int t = threadIdx.x, lane = t & 31, wave = t >> 5;
  const float* dt = bcdt + ((size_t)b * O3 + 2 * St + s) * (size_t)Ll;
  const float* Bm = bcdt + ((size_t)b * O3 + s) * (size_t)Ll;
  float Aa = Av[s];
  float v[16];
  float mx = -INFINITY;
  #pragma unroll
  for (int i = 0; i < 16; ++i) { v[i] = dt[t + i * 256] + Aa; mx = fmaxf(mx, v[i]); }
  for (int o = 16; o > 0; o >>= 1) mx = fmaxf(mx, __shfl_down(mx, o, 32));
  if (lane == 0) red[wave] = mx;
  __syncthreads();
  if (t == 0) {
    float m = red[0];
    for (int i = 1; i < 8; ++i) m = fmaxf(m, red[i]);
    bc = m;
  }
  __syncthreads();
  mx = bc;
  float sum = 0.f;
  #pragma unroll
  for (int i = 0; i < 16; ++i) { v[i] = __expf(v[i] - mx); sum += v[i]; }
  for (int o = 16; o > 0; o >>= 1) sum += __shfl_down(sum, o, 32);
  if (lane == 0) red[wave] = sum;
  __syncthreads();
  if (t == 0) {
    float m = 0.f;
    for (int i = 0; i < 8; ++i) m += red[i];
    bc = m;
  }
  __syncthreads();
  float inv = 1.f / bc;
  bf16_t* abp = ab + ((size_t)b * St + s) * (size_t)Ll;
  #pragma unroll
  for (int i = 0; i < 16; ++i)
    abp[t + i * 256] = (bf16_t)(v[i] * inv * Bm[t + i * 256]);
}

// ---------------------------------------------------------------------------
// hg^T bf16 [b][s][c] from hz fp32 [b][512][64]
// ---------------------------------------------------------------------------
__global__ void k_hz_gate(const float* __restrict__ hz, bf16_t* __restrict__ hgt) {
  int i = blockIdx.x * 256 + threadIdx.x;  // b*64*256
  int b = i >> 14, r = i & 16383, s = r >> 8, c = r & 255;
  float hp = hz[((size_t)b * 512 + c) * St + s];
  float z  = hz[((size_t)b * 512 + 256 + c) * St + s];
  hgt[i] = (bf16_t)(hp * z * sigm_(z));
}

// ---------------------------------------------------------------------------
// x2 = (1-a1)*x1 + a1*(yraw + xn*Dp)
// ---------------------------------------------------------------------------
__global__ void k_y_epi(const float* __restrict__ x1, const float* __restrict__ xn,
                        const float* __restrict__ yraw, const float* __restrict__ Dp,
                        const float* __restrict__ al, float* __restrict__ x2) {
  int idx = blockIdx.x * 256 + threadIdx.x;
  int b = blockIdx.z;
  int c = idx >> 12;
  size_t off = (size_t)b * Cd * (size_t)Ll + idx;
  float a = sigm_(al[c]);
  float y = yraw[off] + xn[off] * Dp[c];
  x2[off] = (1.f - a) * x1[off] + a * y;
}

// ---------------------------------------------------------------------------
// FFN mid: f1 fp32 [b][l][1024] -> LN(fc1_ln) -> SiLU -> bf16 [b][l][1024]
// one block per pixel row
// ---------------------------------------------------------------------------
__global__ void __launch_bounds__(256)
k_ffn_ln_silu(const float* __restrict__ f1, const float* __restrict__ lw,
              const float* __restrict__ lb, bf16_t* __restrict__ out) {
  __shared__ float r1[8], r2[8];
  __shared__ float sm, sr;
  const float* p = f1 + (size_t)blockIdx.x * HID;
  int t = threadIdx.x, lane = t & 31, wave = t >> 5;
  float v[4], s1 = 0.f, s2 = 0.f;
  #pragma unroll
  for (int i = 0; i < 4; ++i) { v[i] = p[t + i * 256]; s1 += v[i]; s2 += v[i] * v[i]; }
  for (int o = 16; o > 0; o >>= 1) { s1 += __shfl_down(s1, o, 32); s2 += __shfl_down(s2, o, 32); }
  if (lane == 0) { r1[wave] = s1; r2[wave] = s2; }
  __syncthreads();
  if (t == 0) {
    float a = 0.f, q = 0.f;
    for (int i = 0; i < 8; ++i) { a += r1[i]; q += r2[i]; }
    float m = a / (float)HID;
    sm = m;
    sr = rsqrtf(q / (float)HID - m * m + EPS);
  }
  __syncthreads();
  bf16_t* o = out + (size_t)blockIdx.x * HID;
  float m = sm, rs = sr;
  #pragma unroll
  for (int i = 0; i < 4; ++i) {
    int c = t + i * 256;
    float n = lw[c] * (v[i] - m) * rs + lb[c];
    o[c] = (bf16_t)(n * sigm_(n));
  }
}

// ---------------------------------------------------------------------------
// final: f2raw fp32 [b][l][256] -> LN(fc2_ln); x4 = (1-a3)*x3 + a3*f2, NCHW out
// LDS transpose, 32 pixels per block. grid(Ll/32, 1, Bn)
// ---------------------------------------------------------------------------
__global__ void __launch_bounds__(256)
k_final(const float* __restrict__ f2raw, const float* __restrict__ lw,
        const float* __restrict__ lb, const float* __restrict__ x3,
        const float* __restrict__ al, float* __restrict__ out) {
  __shared__ float tile[32 * 257];
  __shared__ float smean[32], srstd[32];
  int b = blockIdx.z;
  int l0 = blockIdx.x * 32;
  for (int i = threadIdx.x; i < 32 * 256; i += 256) {
    int p = i >> 8, c = i & 255;
    tile[p * 257 + c] = f2raw[((size_t)b * Ll + l0 + p) * (size_t)Cd + c];
  }
  __syncthreads();
  if (threadIdx.x < 32) {
    int p = threadIdx.x;
    float s1 = 0.f, s2 = 0.f;
    for (int c = 0; c < 256; ++c) { float v = tile[p * 257 + c]; s1 += v; s2 += v * v; }
    float m = s1 * (1.f / 256.f);
    smean[p] = m;
    srstd[p] = rsqrtf(s2 * (1.f / 256.f) - m * m + EPS);
  }
  __syncthreads();
  int p = threadIdx.x & 31;
  for (int c = (threadIdx.x >> 5); c < 256; c += 8) {
    float n = lw[c] * (tile[p * 257 + c] - smean[p]) * srstd[p] + lb[c];
    float a = sigm_(al[c]);
    size_t off = ((size_t)b * Cd + c) * (size_t)Ll + l0 + p;
    out[off] = (1.f - a) * x3[off] + a * n;
  }
}

// ===========================================================================
extern "C" void kernel_launch(void* const* d_in, const int* in_sizes, int n_in,
                              void* d_out, int out_size, void* d_ws, size_t ws_size,
                              hipStream_t stream) {
  const float* x      = (const float*)d_in[0];
  const float* alpha  = (const float*)d_in[1];
  const float* ln_w   = (const float*)d_in[2];
  const float* ln_b   = (const float*)d_in[3];
  const float* dw1_w  = (const float*)d_in[4];
  const float* dw1_lw = (const float*)d_in[5];
  const float* dw1_lb = (const float*)d_in[6];
  const float* dw2_w  = (const float*)d_in[7];
  const float* dw2_lw = (const float*)d_in[8];
  const float* dw2_lb = (const float*)d_in[9];
  const float* bcdt_w = (const float*)d_in[10];
  const float* bcdt_b = (const float*)d_in[11];
  const float* dwm_w  = (const float*)d_in[12];
  const float* dwm_b  = (const float*)d_in[13];
  const float* hz_w   = (const float*)d_in[14];
  const float* hz_b   = (const float*)d_in[15];
  const float* outp_w = (const float*)d_in[16];
  const float* outp_b = (const float*)d_in[17];
  const float* Avec   = (const float*)d_in[18];
  const float* Dp     = (const float*)d_in[19];
  const float* fc1_w  = (const float*)d_in[20];
  const float* fc1_lw = (const float*)d_in[21];
  const float* fc1_lb = (const float*)d_in[22];
  const float* fc2_w  = (const float*)d_in[23];
  const float* fc2_lw = (const float*)d_in[24];
  const float* fc2_lb = (const float*)d_in[25];

  constexpr size_t F = (size_t)Bn * Cd * Ll;   // 16,777,216 elems
  char* ws = (char*)d_ws;
  size_t off = 0;
  auto alloc = [&](size_t bytes) { size_t o = off; off += (bytes + 255) & ~255ull; return o; };

  float*  x1     = (float*)(ws + alloc(F * 4));               // later aliases x3
  float*  xn     = (float*)(ws + alloc(F * 4));
  float*  c1     = (float*)(ws + alloc(F * 4));               // conv scratch, later f2raw
  float*  x2     = (float*)(ws + alloc(F * 4));
  float*  yraw   = (float*)(ws + alloc(F * 4));
  float*  mean   = (float*)(ws + alloc((size_t)Bn * Ll * 4));
  float*  rstd   = (float*)(ws + alloc((size_t)Bn * Ll * 4));
  float*  bcdt0  = (float*)(ws + alloc((size_t)Bn * O3 * Ll * 4));
  float*  bcdt1  = (float*)(ws + alloc((size_t)Bn * O3 * Ll * 4));
  float*  f1     = (float*)(ws + alloc((size_t)Bn * Ll * HID * 4));
  float*  hzf    = (float*)(ws + alloc((size_t)Bn * 512 * St * 4));
  float*  hof    = (float*)(ws + alloc((size_t)Bn * Cd * St * 4));
  bf16_t* xn_cl  = (bf16_t*)(ws + alloc(F * 2));
  bf16_t* xn_lc  = (bf16_t*)(ws + alloc(F * 2));
  bf16_t* x3_lc  = (bf16_t*)(ws + alloc(F * 2));
  bf16_t* f1n    = (bf16_t*)(ws + alloc((size_t)Bn * Ll * HID * 2));
  bf16_t* AB     = (bf16_t*)(ws + alloc((size_t)Bn * St * Ll * 2));
  bf16_t* CmT    = (bf16_t*)(ws + alloc((size_t)Bn * Ll * St * 2));
  bf16_t* hT     = (bf16_t*)(ws + alloc((size_t)Bn * St * Cd * 2));
  bf16_t* hgT    = (bf16_t*)(ws + alloc((size_t)Bn * St * Cd * 2));
  bf16_t* ho_bf  = (bf16_t*)(ws + alloc((size_t)Bn * Cd * St * 2));
  bf16_t* w_bcdt = (bf16_t*)(ws + alloc((size_t)O3 * Cd * 2));
  bf16_t* w_hz   = (bf16_t*)(ws + alloc((size_t)512 * Cd * 2));
  bf16_t* w_outp = (bf16_t*)(ws + alloc((size_t)Cd * Cd * 2));
  bf16_t* w_fc1  = (bf16_t*)(ws + alloc((size_t)HID * Cd * 2));
  bf16_t* w_fc2  = (bf16_t*)(ws + alloc((size_t)Cd * HID * 2));
  float*  x3     = x1;                                        // alias (x1 dead after k_y_epi)
  float*  f2raw  = c1;                                        // alias (c1 dead after x3)
  float*  x4_out = (float*)d_out;
  float*  h_out  = (float*)d_out + F;                         // [b][256][64]

  // --- weight packs ---
  k_cvt_bf16<<<(O3 * Cd + 255) / 256, 256, 0, stream>>>(bcdt_w, w_bcdt, O3 * Cd);
  k_cvt_bf16<<<(512 * Cd + 255) / 256, 256, 0, stream>>>(hz_w, w_hz, 512 * Cd);
  k_cvt_bf16<<<(Cd * Cd + 255) / 256, 256, 0, stream>>>(outp_w, w_outp, Cd * Cd);
  k_cvt_bf16<<<(HID * Cd + 255) / 256, 256, 0, stream>>>(fc1_w, w_fc1, HID * Cd);
  k_cvt_bf16<<<(Cd * HID + 255) / 256, 256, 0, stream>>>(fc2_w, w_fc2, Cd * HID);

  dim3 ew(Cd * Ll / 256, 1, Bn);
  dim3 st(Bn * Ll / 256);

  // --- sub-layer 1: x1 = gate(x, LN(dwconv(x))) ---
  k_dwconv3x3<<<dim3(Cd * Ll / 256, 1, Bn), 256, 0, stream>>>(x, dw1_w, nullptr, c1, Cd);
  k_chan_stats<<<st, 256, 0, stream>>>(c1, mean, rstd, Cd);
  k_ln_gate<<<ew, 256, 0, stream>>>(x, c1, mean, rstd, dw1_lw, dw1_lb, alpha + 0 * Cd, x1);

  // --- xn = LN(x1) (+ bf16 packs in both layouts) ---
  k_chan_stats<<<st, 256, 0, stream>>>(x1, mean, rstd, Cd);
  k_ln_pack<<<ew, 256, 0, stream>>>(x1, mean, rstd, ln_w, ln_b, xn, xn_cl);
  k_transpack<<<dim3(Ll / 32, Cd / 32, Bn), 256, 0, stream>>>(xn, xn_lc, Cd, Ll,
                                                             (size_t)Cd * Ll, (size_t)Ll * Cd);

  // --- bcdt = W_bcdt @ xn + b : M=192,N=4096,K=256 ---
  k_gemm_bf16<<<dim3(Ll / 64, O3 / 64, Bn), 128, 0, stream>>>(
      w_bcdt, xn_lc, bcdt0, bcdt_b, O3, Ll, Cd, 0, (size_t)Ll * Cd, (size_t)O3 * Ll);
  k_dwconv3x3<<<dim3(O3 * Ll / 256, 1, Bn), 256, 0, stream>>>(bcdt0, dwm_w, dwm_b, bcdt1, O3);

  // --- softmax(dt+A)*B -> AB bf16 [s][l]; Cm -> bf16 [l][s] ---
  k_softmax_ab<<<dim3(St, 1, Bn), 256, 0, stream>>>(bcdt1, Avec, AB);
  k_transpack<<<dim3(Ll / 32, St / 32, Bn), 256, 0, stream>>>(bcdt1 + (size_t)St * Ll, CmT,
                                                             St, Ll, (size_t)O3 * Ll, (size_t)Ll * St);

  // --- h = xn @ AB^T : M=256,N=64,K=4096 -> straight into d_out h slot ---
  k_gemm_bf16<<<dim3(St / 64, Cd / 64, Bn), 128, 0, stream>>>(
      xn_cl, AB, h_out, nullptr, Cd, St, Ll, (size_t)Cd * Ll, (size_t)St * Ll, (size_t)Cd * St);
  k_transpack<<<dim3(St / 32, Cd / 32, Bn), 256, 0, stream>>>(h_out, hT, Cd, St,
                                                             (size_t)Cd * St, (size_t)St * Cd);

  // --- hz = W_hz @ h + b : M=512,N=64,K=256; gate; ho = W_outp @ hg + b ---
  k_gemm_bf16<<<dim3(St / 64, 512 / 64, Bn), 128, 0, stream>>>(
      w_hz, hT, hzf, hz_b, 512, St, Cd, 0, (size_t)St * Cd, (size_t)512 * St);
  k_hz_gate<<<(Bn * St * Cd) / 256, 256, 0, stream>>>(hzf, hgT);
  k_gemm_bf16<<<dim3(St / 64, Cd / 64, Bn), 128, 0, stream>>>(
      w_outp, hgT, hof, outp_b, Cd, St, Cd, 0, (size_t)St * Cd, (size_t)Cd * St);
  k_cvt_bf16<<<(Bn * Cd * St) / 256, 256, 0, stream>>>(hof, ho_bf, Bn * Cd * St);

  // --- y = ho @ Cm : M=256,N=4096,K=64 ; x2 = lerp(x1, y + xn*Dp) ---
  k_gemm_bf16<<<dim3(Ll / 64, Cd / 64, Bn), 128, 0, stream>>>(
      ho_bf, CmT, yraw, nullptr, Cd, Ll, St, (size_t)Cd * St, (size_t)Ll * St, (size_t)Cd * Ll);
  k_y_epi<<<ew, 256, 0, stream>>>(x1, xn, yraw, Dp, alpha + 1 * Cd, x2);

  // --- sub-layer 3: x3 = gate(x2, LN(dwconv(x2))) ---
  k_dwconv3x3<<<dim3(Cd * Ll / 256, 1, Bn), 256, 0, stream>>>(x2, dw2_w, nullptr, c1, Cd);
  k_chan_stats<<<st, 256, 0, stream>>>(c1, mean, rstd, Cd);
  k_ln_gate<<<ew, 256, 0, stream>>>(x2, c1, mean, rstd, dw2_lw, dw2_lb, alpha + 2 * Cd, x3);
  k_transpack<<<dim3(Ll / 32, Cd / 32, Bn), 256, 0, stream>>>(x3, x3_lc, Cd, Ll,
                                                             (size_t)Cd * Ll, (size_t)Ll * Cd);

  // --- FFN: f1 = x3 @ fc1^T (pixel-major M=L,N=1024,K=256) ---
  k_gemm_bf16<<<dim3(HID / 64, Ll / 64, Bn), 128, 0, stream>>>(
      x3_lc, w_fc1, f1, nullptr, Ll, HID, Cd, (size_t)Ll * Cd, 0, (size_t)Ll * HID);
  k_ffn_ln_silu<<<Bn * Ll, 256, 0, stream>>>(f1, fc1_lw, fc1_lb, f1n);
  // f2raw = f1n @ fc2^T : M=L,N=256,K=1024
  k_gemm_bf16<<<dim3(Cd / 64, Ll / 64, Bn), 128, 0, stream>>>(
      f1n, w_fc2, f2raw, nullptr, Ll, Cd, HID, (size_t)Ll * HID, 0, (size_t)Ll * Cd);

  // --- final LN + lerp -> NCHW x4 ---
  k_final<<<dim3(Ll / 32, 1, Bn), 256, 0, stream>>>(f2raw, fc2_lw, fc2_lb, x3,
                                                   alpha + 3 * Cd, x4_out);
  (void)in_sizes; (void)n_in; (void)out_size; (void)ws_size;
}
